// GATLayer_79757542687072
// MI455X (gfx1250) — compile-verified, matrix operand order checked by reading, not run on hardware
//
#include <hip/hip_runtime.h>

// ---------------------------------------------------------------------------
// GAT layer for MI455X (gfx1250, wave32).
//   N=4096 nodes, IN_DIM=512, H=8 heads, D=64.
// Pipeline:
//   1. cvt_f32_f16    : data -> Xh, W -> Wh             (f16 copies, L2-resident)
//   2. gemm_proj      : Yh = f16(Xh @ Wh^T + b)         (v_wmma_f32_16x16x32_f16)
//   3. proj_scores_T  : srcT/trgT[h][n] = log2e * <Yh[n,h,:], proj[h,:]>
//   4. head_max       : per-head max of srcT / trgT (static softmax shift)
//   5. build_vfrag    : pre-swizzle Yh into WMMA B-fragment order (one pass)
//   6. gat_attention  : static-shift softmax attention; conn streamed ONCE via
//                       double-buffered global_load_async_to_lds_b128; P tile
//                       built directly in A-fragment layout; P@V and row sums
//                       (ones-column trick) by WMMA.
// ---------------------------------------------------------------------------

typedef _Float16 v16h __attribute__((ext_vector_type(16)));
typedef _Float16 v8h  __attribute__((ext_vector_type(8)));
typedef __fp16   v2fp __attribute__((ext_vector_type(2)));
typedef float    v8f  __attribute__((ext_vector_type(8)));

#define NN      4096
#define IND     512
#define HD      512
#define NHEAD   8
#define DHEAD   64
#define LOG2E   1.44269504088896340736f

// ------------------------------------------------------------- small helpers
__device__ __forceinline__ float exp2_fast(float x) {
#if __has_builtin(__builtin_amdgcn_exp2f)
  return __builtin_amdgcn_exp2f(x);          // v_exp_f32
#else
  return exp2f(x);
#endif
}

__device__ __forceinline__ unsigned lds_off(const void* p) {
  // flat LDS aperture keeps the LDS byte offset in addr[31:0] (ISA 10.2)
  return (unsigned)(uintptr_t)p;
}

// CDNA5 async copy: global -> LDS, 16B per lane, tracked by ASYNCcnt.
__device__ __forceinline__ void async_copy_b128(unsigned lds_dst, const float* g) {
  asm volatile("global_load_async_to_lds_b128 %0, %1, off"
               :: "v"(lds_dst), "v"((unsigned long long)(uintptr_t)g)
               : "memory");
}
__device__ __forceinline__ void wait_async0() {
  asm volatile("s_wait_asynccnt 0" ::: "memory");
}

// --------------------------------------------------------------- conversions
__global__ void cvt_f32_f16(const float* __restrict__ src,
                            _Float16* __restrict__ dst, int n) {
  int i = blockIdx.x * blockDim.x + threadIdx.x;
  if (i < n) dst[i] = (_Float16)src[i];
}

// ------------------------------------------------------------ projection GEMM
__global__ __launch_bounds__(256) void gemm_proj(
    const _Float16* __restrict__ Xh, const _Float16* __restrict__ Wh,
    const float* __restrict__ bias, _Float16* __restrict__ Yh) {
  const int lane = threadIdx.x & 31;
  const int wave = threadIdx.x >> 5;
  const int tile = blockIdx.x * 8 + wave;
  const int row0 = (tile >> 5) << 4;
  const int col0 = (tile & 31) << 4;
  const int r = lane & 15, hf = lane >> 4;

  v8f acc = {0.f, 0.f, 0.f, 0.f, 0.f, 0.f, 0.f, 0.f};
  const _Float16* arow = Xh + (size_t)(row0 + r) * IND + (hf << 3);
  const _Float16* bcol = Wh + (size_t)(col0 + r) * IND + (hf << 3);

  for (int kt = 0; kt < IND; kt += 32) {
    union { v16h v; v8h p[2]; } a, b;
    a.p[0] = *(const v8h*)(arow + kt);
    a.p[1] = *(const v8h*)(arow + kt + 16);
    b.p[0] = *(const v8h*)(bcol + kt);
    b.p[1] = *(const v8h*)(bcol + kt + 16);
    acc = __builtin_amdgcn_wmma_f32_16x16x32_f16(false, a.v, false, b.v,
                                                 (short)0, acc, false, false);
  }
  const int c  = col0 + r;
  const float bv = bias[c];
#pragma unroll
  for (int i = 0; i < 8; ++i) {
    int rr = row0 + (hf << 3) + i;
    Yh[(size_t)rr * HD + c] = (_Float16)(acc[i] + bv);
  }
}

// ---------------------------------------- per-node logits, transposed + scaled
__global__ void proj_scores_T(const _Float16* __restrict__ Yh,
                              const float* __restrict__ sproj,
                              const float* __restrict__ tproj,
                              float* __restrict__ srcT, float* __restrict__ trgT) {
  int i = blockIdx.x * blockDim.x + threadIdx.x;   // over N*H
  if (i >= NN * NHEAD) return;
  int n = i >> 3, h = i & 7;
  const _Float16* y = Yh + (size_t)n * HD + h * DHEAD;
  const float* sp = sproj + h * DHEAD;
  const float* tp = tproj + h * DHEAD;
  float s = 0.f, t = 0.f;
#pragma unroll 8
  for (int d = 0; d < DHEAD; ++d) {
    float v = (float)y[d];
    s += v * sp[d];
    t += v * tp[d];
  }
  srcT[(size_t)h * NN + n] = s * LOG2E;   // log2-domain
  trgT[(size_t)h * NN + n] = t * LOG2E;
}

// ------------------------------------------------- per-head maxima (16 waves)
__global__ void head_max(const float* __restrict__ srcT,
                         const float* __restrict__ trgT,
                         float* __restrict__ smax, float* __restrict__ tmax) {
  int b = blockIdx.x;                 // 0..7: src head b ; 8..15: trg head b-8
  int lane = threadIdx.x;             // 32 threads
  const float* base = (b < 8) ? (srcT + (size_t)b * NN)
                              : (trgT + (size_t)(b - 8) * NN);
  float m = -3.0e38f;
  for (int i = lane; i < NN; i += 32) m = fmaxf(m, base[i]);
#pragma unroll
  for (int off = 16; off; off >>= 1) m = fmaxf(m, __shfl_xor(m, off, 32));
  if (lane == 0) { if (b < 8) smax[b] = m; else tmax[b - 8] = m; }
}

// -------------------------------------------- pre-swizzle V to B-fragment order
__global__ void build_vfrag(const _Float16* __restrict__ Yh,
                            _Float16* __restrict__ Vfrag) {
  int gid = blockIdx.x * blockDim.x + threadIdx.x;
  if (gid >= (NN / 32) * NHEAD * 4 * 32) return;
  int lane = gid & 31, sub = (gid >> 5) & 3, h = (gid >> 7) & 7, mtile = gid >> 10;
  int hf = lane >> 4, cl = lane & 15;
  int m0 = mtile * 32;
  const _Float16* ysrc = Yh + h * DHEAD + sub * 16 + cl;
  v16h v;
#pragma unroll
  for (int e = 0; e < 16; ++e) {
    int k = e + (e & 8) + (hf << 3);
    v[e] = ysrc[(size_t)(m0 + k) * HD];
  }
  *(v16h*)(Vfrag + (size_t)gid * 16) = v;
}

// ----------------------------------------------------------- fused attention
// Block = 256 threads = 8 waves = 8 heads; block owns 16 query rows.
// Static softmax shift M_h (valid since conn<=0, leaky monotonic) removes all
// online-softmax bookkeeping. conn/trg tiles arrive via async-to-LDS DMA,
// double-buffered so the copy overlaps score computation + WMMA.
__global__ __launch_bounds__(256) void gat_attention(
    const float* __restrict__ conn, const float* __restrict__ srcT,
    const float* __restrict__ trgT, const float* __restrict__ smax,
    const float* __restrict__ tmax, const _Float16* __restrict__ Vfrag,
    float* __restrict__ out) {
  __shared__ __align__(16) float conn_s[2][16 * 32];
  __shared__ __align__(16) float trg_s[2][NHEAD * 32];

  const int tid  = threadIdx.x;
  const int lane = tid & 31;
  const int h    = tid >> 5;            // wave id == head
  const int r    = lane & 15;           // A-fragment row
  const int hf   = lane >> 4;           // A-fragment half (selects k group)
  const int n0   = blockIdx.x << 4;     // 16 query rows per block

  const float srcv = srcT[(size_t)h * NN + n0 + r];
  const float negM = -(smax[h] + tmax[h]);      // log2-domain shift

  v8f acc0 = {0.f,0.f,0.f,0.f,0.f,0.f,0.f,0.f};
  v8f acc1 = acc0, acc2 = acc0, acc3 = acc0, acc4 = acc0;

  v16h ones;
#pragma unroll
  for (int e = 0; e < 16; ++e) ones[e] = (_Float16)1.0f;

  // async staging of one (conn 16x32 + trg 8x32) f32 tile: 192 x b128
  auto stage = [&](int m0, int buf) {
    if (tid < 128) {
      int rr = tid >> 3, q = tid & 7;
      async_copy_b128(lds_off(&conn_s[buf][(rr << 5) + (q << 2)]),
                      conn + (size_t)(n0 + rr) * NN + m0 + (q << 2));
    } else if (tid < 192) {
      int u = tid - 128, hs = u >> 3, q = u & 7;
      async_copy_b128(lds_off(&trg_s[buf][(hs << 5) + (q << 2)]),
                      trgT + (size_t)hs * NN + m0 + (q << 2));
    }
  };

  stage(0, 0);
  wait_async0();
  __syncthreads();

  int cur = 0;
  for (int m0 = 0; m0 < NN; m0 += 32) {
    const int nxt = cur ^ 1;
    if (m0 + 32 < NN) stage(m0 + 32, nxt);   // overlap DMA with compute

    // scores in A-fragment layout: element e -> k = e + (e&8) + 8*hf
    float p[16];
#pragma unroll
    for (int e = 0; e < 16; ++e) {
      int k = e + (e & 8) + (hf << 3);
      float t1 = srcv + trg_s[cur][(h << 5) + k];
      float t2 = fmaxf(t1, 0.01f * t1);                       // LeakyReLU
      float arg = fmaf(conn_s[cur][(r << 5) + k], LOG2E, t2 + negM);
      p[e] = exp2_fast(arg);                                  // arg <= 0
    }

    v16h pa;
#if __has_builtin(__builtin_amdgcn_cvt_pkrtz)
    {
      union { v16h v; v2fp h2[8]; } pu;
#pragma unroll
      for (int j = 0; j < 8; ++j)
        pu.h2[j] = __builtin_amdgcn_cvt_pkrtz(p[2 * j], p[2 * j + 1]);
      pa = pu.v;
    }
#else
#pragma unroll
    for (int e = 0; e < 16; ++e) pa[e] = (_Float16)p[e];
#endif

    // P(16x32) @ V(32x64) + row-sum column: 5 WMMAs
    const _Float16* vb = Vfrag + ((size_t)((m0 >> 5) * NHEAD + h) * 4) * 32 * 16;
    v16h b0 = *(const v16h*)(vb + (size_t)(0 * 32 + lane) * 16);
    v16h b1 = *(const v16h*)(vb + (size_t)(1 * 32 + lane) * 16);
    v16h b2 = *(const v16h*)(vb + (size_t)(2 * 32 + lane) * 16);
    v16h b3 = *(const v16h*)(vb + (size_t)(3 * 32 + lane) * 16);
    acc0 = __builtin_amdgcn_wmma_f32_16x16x32_f16(false, pa, false, b0,   (short)0, acc0, false, false);
    acc1 = __builtin_amdgcn_wmma_f32_16x16x32_f16(false, pa, false, b1,   (short)0, acc1, false, false);
    acc2 = __builtin_amdgcn_wmma_f32_16x16x32_f16(false, pa, false, b2,   (short)0, acc2, false, false);
    acc3 = __builtin_amdgcn_wmma_f32_16x16x32_f16(false, pa, false, b3,   (short)0, acc3, false, false);
    acc4 = __builtin_amdgcn_wmma_f32_16x16x32_f16(false, pa, false, ones, (short)0, acc4, false, false);

    wait_async0();        // own async loads for `nxt` done
    __syncthreads();      // everyone done reading `cur` / writing `nxt`
    cur = nxt;
  }

  // normalize: acc4[i] already holds the row sum for row i+8*hf in this lane
#pragma unroll
  for (int i = 0; i < 8; ++i) {
    float inv = 1.0f / acc4[i];
    int rr = n0 + (hf << 3) + i;
    float* o = out + (size_t)rr * HD + h * DHEAD + r;
    o[0]  = acc0[i] * inv;
    o[16] = acc1[i] * inv;
    o[32] = acc2[i] * inv;
    o[48] = acc3[i] * inv;
  }
}

// ---------------------------------------------------------------------------
extern "C" void kernel_launch(void* const* d_in, const int* in_sizes, int n_in,
                              void* d_out, int out_size, void* d_ws, size_t ws_size,
                              hipStream_t stream) {
  const float* data  = (const float*)d_in[0];
  const float* conn  = (const float*)d_in[1];
  const float* W     = (const float*)d_in[2];
  const float* bias  = (const float*)d_in[3];
  const float* sproj = (const float*)d_in[4];
  const float* tproj = (const float*)d_in[5];
  float* out = (float*)d_out;

  // workspace layout (~12.75 MB, all L2-resident)
  char* ws = (char*)d_ws;
  _Float16* Xh  = (_Float16*)(ws);                                     // 4   MB
  _Float16* Wh  = (_Float16*)(ws + (size_t)(4u << 20));                // 0.5 MB
  _Float16* Yh  = (_Float16*)(ws + (size_t)(4u << 20) + (512u << 10)); // 4   MB
  _Float16* Vf  = (_Float16*)(ws + (size_t)(8u << 20) + (512u << 10)); // 4   MB
  float*    srcT = (float*)(ws + (size_t)(12u << 20) + (512u << 10));  // 128 KB
  float*    trgT = srcT + NN * NHEAD;                                  // 128 KB
  float*    smax = trgT + NN * NHEAD;                                  // 32 B
  float*    tmax = smax + NHEAD;                                       // 32 B

  cvt_f32_f16<<<(NN * IND) / 256, 256, 0, stream>>>(data, Xh, NN * IND);
  cvt_f32_f16<<<(HD * IND) / 256, 256, 0, stream>>>(W, Wh, HD * IND);
  gemm_proj<<<(NN / 16) * (HD / 16) / 8, 256, 0, stream>>>(Xh, Wh, bias, Yh);
  proj_scores_T<<<(NN * NHEAD) / 256, 256, 0, stream>>>(Yh, sproj, tproj, srcT, trgT);
  head_max<<<16, 32, 0, stream>>>(srcT, trgT, smax, tmax);
  build_vfrag<<<((NN / 32) * NHEAD * 4 * 32) / 256, 256, 0, stream>>>(Yh, Vf);
  gat_attention<<<NN / 16, 256, 0, stream>>>(conn, srcT, trgT, smax, tmax, Vf, out);
}
